// DynamicMultiHeadAttention_44452911513816
// MI455X (gfx1250) — matrix operand order repaired and checked by reading, not compiled, not run on hardware
//
#include <hip/hip_runtime.h>
#include <hip/hip_bf16.h>
#include <math.h>

// ---------------- problem constants ----------------
constexpr int E  = 1024;          // embed dim (= N = K of all GEMMs)
constexpr int H  = 16;            // heads
constexpr int Dh = 64;            // head dim
constexpr int Bq = 4;             // batch
constexpr int S  = 2048;          // sequence
constexpr int Mrows = Bq * S;     // 8192 rows for projection GEMMs

// ---------------- workspace layout (bytes) ----------------
constexpr size_t BETAS_OFF = 0;                                    // B*H floats (pad 1KB)
constexpr size_t HALF_MAT  = (size_t)Mrows * E * sizeof(_Float16); // 16 MB
constexpr size_t QH_OFF = 1024;
constexpr size_t KH_OFF = QH_OFF + HALF_MAT;
constexpr size_t VH_OFF = KH_OFF + HALF_MAT;
constexpr size_t AH_OFF = VH_OFF + HALF_MAT;

// ---------------- vector types ----------------
typedef __attribute__((ext_vector_type(16))) _Float16 v16h;
typedef __attribute__((ext_vector_type(8)))  _Float16 v8h;
typedef __attribute__((ext_vector_type(8)))  float    v8f;

__device__ __forceinline__ v16h mk16(const _Float16* lo, const _Float16* hi) {
    union { v16h v; v8h p[2]; } u;
    u.p[0] = *(const v8h*)lo;
    u.p[1] = *(const v8h*)hi;
    return u.v;
}

// f32 -> f16 (or f16 passthrough) 4-element tile loader into LDS
__device__ __forceinline__ void cvt_store4(const float* src, _Float16* dst) {
    float4 f = *(const float4*)src;
    dst[0] = (_Float16)f.x; dst[1] = (_Float16)f.y;
    dst[2] = (_Float16)f.z; dst[3] = (_Float16)f.w;
}
__device__ __forceinline__ void cvt_store4(const _Float16* src, _Float16* dst) {
    *(uint2*)dst = *(const uint2*)src;   // 4 halves = 8 bytes
}

// ---------------- DPP16 cross-lane reductions (within 16-lane groups) ----
// ctrl: 0xB1 quad_perm(1,0,3,2)=xor1, 0x4E quad_perm(2,3,0,1)=xor2,
//       0x141 row_half_mirror (== xor4 once quads uniform),
//       0x140 row_mirror      (== xor8 once octs uniform)
template <int CTRL>
__device__ __forceinline__ float dpp_f(float x) {
    return __builtin_bit_cast(float,
        __builtin_amdgcn_update_dpp(0, __builtin_bit_cast(int, x),
                                    CTRL, 0xF, 0xF, true));
}
__device__ __forceinline__ float rowmax16(float x) {
    x = fmaxf(x, dpp_f<0xB1>(x));
    x = fmaxf(x, dpp_f<0x4E>(x));
    x = fmaxf(x, dpp_f<0x141>(x));
    x = fmaxf(x, dpp_f<0x140>(x));
    return x;
}
__device__ __forceinline__ float rowsum16(float x) {
    x += dpp_f<0xB1>(x);
    x += dpp_f<0x4E>(x);
    x += dpp_f<0x141>(x);
    x += dpp_f<0x140>(x);
    return x;
}

// =====================================================================
// Kernel 1: pooled mean + betas = sigmoid(pooled @ Wb^T + bb)
// =====================================================================
__global__ __launch_bounds__(256)
void betas_kernel(const float* __restrict__ q, const float* __restrict__ Wb,
                  const float* __restrict__ bb, float* __restrict__ betas) {
    __shared__ float pooled[E];
    const int b = blockIdx.x, t = threadIdx.x;
    const float* qb = q + (size_t)b * S * E;
    float s0 = 0.f, s1 = 0.f, s2 = 0.f, s3 = 0.f;
    for (int s = 0; s < S; ++s) {
        const float* r = qb + (size_t)s * E;
        s0 += r[t]; s1 += r[t + 256]; s2 += r[t + 512]; s3 += r[t + 768];
    }
    const float inv = 1.0f / (float)S;
    pooled[t] = s0 * inv; pooled[t + 256] = s1 * inv;
    pooled[t + 512] = s2 * inv; pooled[t + 768] = s3 * inv;
    __syncthreads();
    if (t < H) {
        float acc = bb[t];
        const float* wr = Wb + (size_t)t * E;
        for (int e = 0; e < E; ++e) acc += pooled[e] * wr[e];
        betas[b * H + t] = 1.0f / (1.0f + __expf(-acc));
    }
}

// =====================================================================
// Kernel 2/4: C[M x E] = A[M x E] * W[E x E]^T + bias   (WMMA f16)
//   block tile 128x128, 8 waves in 2x4, each wave 64x32 (4x2 WMMA tiles)
// =====================================================================
template <typename TA, typename TOUT>
__global__ __launch_bounds__(256)
void gemm_wt_kernel(const TA* __restrict__ A, const float* __restrict__ W,
                    const float* __restrict__ bias, TOUT* __restrict__ Cout) {
    constexpr int LDT = 40;                       // LDS row stride in halves
    __shared__ _Float16 As[128 * LDT];
    __shared__ _Float16 Ws[128 * LDT];

    const int tid  = threadIdx.x;
    const int lane = tid & 31;
    const int wv   = tid >> 5;        // 0..7
    const int wm   = wv >> 2;         // 0..1  (64 rows each)
    const int wn   = wv & 3;          // 0..3  (32 cols each)
    const int m0   = blockIdx.y * 128;
    const int n0   = blockIdx.x * 128;
    const int ln16 = lane & 15;
    const int kbA  = (lane < 16) ? 0 : 8;    // A-frag k base
    const int kbB  = (lane < 16) ? 0 : 8;    // B-frag k base

    v8f acc[4][2];
#pragma unroll
    for (int i = 0; i < 4; ++i)
#pragma unroll
        for (int j = 0; j < 2; ++j) acc[i][j] = {};

    for (int k0 = 0; k0 < E; k0 += 32) {
        __syncthreads();
#pragma unroll
        for (int it = 0; it < 4; ++it) {
            int id   = tid + it * 256;       // 0..1023
            int row  = id >> 3;              // 0..127
            int cv   = id & 7;               // float4 slot
            cvt_store4(&A[(size_t)(m0 + row) * E + k0 + cv * 4],
                       &As[row * LDT + cv * 4]);
            cvt_store4(&W[(size_t)(n0 + row) * E + k0 + cv * 4],
                       &Ws[row * LDT + cv * 4]);
        }
        __syncthreads();

        v16h af[4], bf[2];
#pragma unroll
        for (int i = 0; i < 4; ++i) {
            const _Float16* p = &As[(wm * 64 + i * 16 + ln16) * LDT + kbA];
            af[i] = mk16(p, p + 16);
        }
#pragma unroll
        for (int j = 0; j < 2; ++j) {
            const _Float16* p = &Ws[(wn * 32 + j * 16 + ln16) * LDT + kbB];
            bf[j] = mk16(p, p + 16);
        }
#pragma unroll
        for (int i = 0; i < 4; ++i)
#pragma unroll
            for (int j = 0; j < 2; ++j)
                acc[i][j] = __builtin_amdgcn_wmma_f32_16x16x32_f16(
                    false, af[i], false, bf[j], (short)0, acc[i][j], false, false);
    }

#pragma unroll
    for (int j = 0; j < 2; ++j) {
        const int col = n0 + wn * 32 + j * 16 + ln16;
        const float bv = bias[col];
#pragma unroll
        for (int i = 0; i < 4; ++i) {
#pragma unroll
            for (int r = 0; r < 8; ++r) {
                const int row = m0 + wm * 64 + i * 16 + r + ((lane < 16) ? 0 : 8);
                Cout[(size_t)row * E + col] = (TOUT)(acc[i][j][r] + bv);
            }
        }
    }
}

// =====================================================================
// Kernel 3: flash attention, beta-scaled, 64-key tiles
//   grid = (S/64, H, B), 128 threads = 4 waves, wave owns 16 q rows
// =====================================================================
__global__ __launch_bounds__(128)
void attn_kernel(const _Float16* __restrict__ Qh, const _Float16* __restrict__ Kh,
                 const _Float16* __restrict__ Vh, const float* __restrict__ betas,
                 _Float16* __restrict__ Ah) {
    constexpr int KT  = 64;   // keys per tile
    constexpr int LDK = 72;   // Ks row stride (halves), 144B
    constexpr int LDV = 72;   // Vst row stride [d][key..64]
    constexpr int LDP = 72;   // Ps row stride  [row][key..64]
    __shared__ _Float16 Ks[KT * LDK];        // [key][d]
    __shared__ _Float16 Vst[Dh * LDV];       // [d][key]  (transposed V tile)
    __shared__ _Float16 Ps[4 * 16 * LDP];    // per-wave P tile [row][key64]

    const int tid  = threadIdx.x;
    const int lane = tid & 31;
    const int wv   = tid >> 5;               // 0..3
    const int ln16 = lane & 15;
    const int b    = blockIdx.z;
    const int h    = blockIdx.y;
    const int q0   = blockIdx.x * 64;
    const float f  = 0.125f * betas[b * H + h];   // (1/sqrt(64)) * beta

    // Q A-fragments (held for whole kernel)
    const _Float16* qrow =
        Qh + ((size_t)(b * S + q0 + wv * 16 + ln16)) * E + h * Dh;
    const int kbA = (lane < 16) ? 0 : 8;     // A-frag k base
    const int kkB = (lane < 16) ? 0 : 8;     // B-frag k base
    const v16h qa0 = mk16(qrow + kbA,      qrow + kbA + 16);
    const v16h qa1 = mk16(qrow + 32 + kbA, qrow + 32 + kbA + 16);

    v8f acc[4]; acc[0] = {}; acc[1] = {}; acc[2] = {}; acc[3] = {};
    float mst[8], lst[8], alr[8];
#pragma unroll
    for (int r = 0; r < 8; ++r) { mst[r] = -1e30f; lst[r] = 0.f; }

    for (int kb = 0; kb < S; kb += KT) {
        __syncthreads();
        // --- stage K tile (64x64) row-major, 16B vectors ---
#pragma unroll
        for (int it = 0; it < 4; ++it) {
            int v   = tid + it * 128;        // 0..511 vec8 slots
            int key = v >> 3, dv = v & 7;
            *(v8h*)&Ks[key * LDK + dv * 8] =
                *(const v8h*)(Kh + ((size_t)(b * S + kb + key)) * E + h * Dh + dv * 8);
        }
        // --- stage V tile transposed (half2 loads): Vst[d][key] ---
#pragma unroll
        for (int it = 0; it < 16; ++it) {
            int v   = tid + it * 128;        // 0..2047 half2 slots
            int key = v >> 5, dp = v & 31;   // dp: d-pair index
            union { unsigned u; _Float16 h[2]; } t;
            t.u = *(const unsigned*)(Vh + ((size_t)(b * S + kb + key)) * E +
                                     h * Dh + dp * 2);
            Vst[(dp * 2 + 0) * LDV + key] = t.h[0];
            Vst[(dp * 2 + 1) * LDV + key] = t.h[1];
        }
        __syncthreads();

        // --- scores: 4 key sub-tiles of 16, each = 2 WMMAs over d ---
        v8f c[4];
#pragma unroll
        for (int jt = 0; jt < 4; ++jt) {
            const _Float16* kp = &Ks[(jt * 16 + ln16) * LDK + kkB];
            v16h kb_lo = mk16(kp,      kp + 16);    // d 0..31
            v16h kb_hi = mk16(kp + 32, kp + 48);    // d 32..63
            v8f s = {};
            s = __builtin_amdgcn_wmma_f32_16x16x32_f16(false, qa0, false, kb_lo,
                                                       (short)0, s, false, false);
            s = __builtin_amdgcn_wmma_f32_16x16x32_f16(false, qa1, false, kb_hi,
                                                       (short)0, s, false, false);
            c[jt] = s;
        }

        // --- online softmax (DPP reductions in 16-lane groups) ---
        _Float16* pw = Ps + wv * 16 * LDP;
#pragma unroll
        for (int r = 0; r < 8; ++r) {
            float x0 = c[0][r] * f, x1 = c[1][r] * f;
            float x2 = c[2][r] * f, x3 = c[3][r] * f;
            float vm = fmaxf(fmaxf(x0, x1), fmaxf(x2, x3));
            vm = rowmax16(vm);
            float mnew  = fmaxf(mst[r], vm);
            float alpha = __expf(mst[r] - mnew);
            float p0 = __expf(x0 - mnew), p1 = __expf(x1 - mnew);
            float p2 = __expf(x2 - mnew), p3 = __expf(x3 - mnew);
            float rs = rowsum16((p0 + p1) + (p2 + p3));
            lst[r] = lst[r] * alpha + rs;
            mst[r] = mnew;
            alr[r] = alpha;
            const int row = r + ((lane < 16) ? 0 : 8);
            pw[row * LDP +      ln16] = (_Float16)p0;
            pw[row * LDP + 16 + ln16] = (_Float16)p1;
            pw[row * LDP + 32 + ln16] = (_Float16)p2;
            pw[row * LDP + 48 + ln16] = (_Float16)p3;
        }
        // cross-lane LDS RAW inside the same wave: drain DS counter
        asm volatile("s_wait_dscnt 0" ::: "memory");

        // --- re-layout P as two A-fragments (16 x 64 keys) ---
        const _Float16* pr = pw + ln16 * LDP + kbA;
        v16h pa0 = mk16(pr,      pr + 16);   // keys 0..31
        v16h pa1 = mk16(pr + 32, pr + 48);   // keys 32..63

        // --- rescale accumulators, then PV WMMAs over 4 d-tiles ---
#pragma unroll
        for (int dt = 0; dt < 4; ++dt) {
#pragma unroll
            for (int r = 0; r < 8; ++r) acc[dt][r] *= alr[r];
            const _Float16* vp = &Vst[(dt * 16 + ln16) * LDV + kkB];
            v16h vb0 = mk16(vp,      vp + 16);   // keys 0..31
            v16h vb1 = mk16(vp + 32, vp + 48);   // keys 32..63
            acc[dt] = __builtin_amdgcn_wmma_f32_16x16x32_f16(
                false, pa0, false, vb0, (short)0, acc[dt], false, false);
            acc[dt] = __builtin_amdgcn_wmma_f32_16x16x32_f16(
                false, pa1, false, vb1, (short)0, acc[dt], false, false);
        }
        __syncthreads();
    }

    // --- normalize and store (merged back to [b][s][h*64+d], f16) ---
#pragma unroll
    for (int r = 0; r < 8; ++r) {
        const float invl = 1.0f / lst[r];
        const int s = q0 + wv * 16 + r + ((lane < 16) ? 0 : 8);
        _Float16* orow = Ah + ((size_t)(b * S + s)) * E + h * Dh;
#pragma unroll
        for (int dt = 0; dt < 4; ++dt)
            orow[dt * 16 + ln16] = (_Float16)(acc[dt][r] * invl);
    }
}

// =====================================================================
// launcher
// =====================================================================
extern "C" void kernel_launch(void* const* d_in, const int* in_sizes, int n_in,
                              void* d_out, int out_size, void* d_ws, size_t ws_size,
                              hipStream_t stream) {
    const float* query = (const float*)d_in[0];
    const float* Wq    = (const float*)d_in[1];
    const float* bq    = (const float*)d_in[2];
    const float* Wk    = (const float*)d_in[3];
    const float* bk    = (const float*)d_in[4];
    const float* Wv    = (const float*)d_in[5];
    const float* bv    = (const float*)d_in[6];
    const float* Wo    = (const float*)d_in[7];
    const float* bo    = (const float*)d_in[8];
    const float* Wb    = (const float*)d_in[9];
    const float* bb    = (const float*)d_in[10];

    float*     betas = (float*)((char*)d_ws + BETAS_OFF);
    _Float16*  Qh    = (_Float16*)((char*)d_ws + QH_OFF);
    _Float16*  Kh    = (_Float16*)((char*)d_ws + KH_OFF);
    _Float16*  Vh    = (_Float16*)((char*)d_ws + VH_OFF);
    _Float16*  Ah    = (_Float16*)((char*)d_ws + AH_OFF);
    float*     out   = (float*)d_out;

    // 1) betas
    betas_kernel<<<Bq, 256, 0, stream>>>(query, Wb, bb, betas);

    // 2) Q/K/V projections (f32 in, f16 out)
    dim3 ggrid(E / 128, Mrows / 128);
    gemm_wt_kernel<float, _Float16><<<ggrid, 256, 0, stream>>>(query, Wq, bq, Qh);
    gemm_wt_kernel<float, _Float16><<<ggrid, 256, 0, stream>>>(query, Wk, bk, Kh);
    gemm_wt_kernel<float, _Float16><<<ggrid, 256, 0, stream>>>(query, Wv, bv, Vh);

    // 3) beta-scaled flash attention
    attn_kernel<<<dim3(S / 64, H, Bq), 128, 0, stream>>>(Qh, Kh, Vh, betas, Ah);

    // 4) output projection (f16 in, f32 out)
    gemm_wt_kernel<_Float16, float><<<ggrid, 256, 0, stream>>>(Ah, Wo, bo, out);
}
